// Encoder_68015102099714
// MI455X (gfx1250) — compile-verified, matrix operand order checked by reading, not run on hardware
//
#include <hip/hip_runtime.h>
#include <hip/hip_bf16.h>

typedef _Float16 h8   __attribute__((ext_vector_type(8)));
typedef _Float16 v16h __attribute__((ext_vector_type(16)));
typedef float    v8f  __attribute__((ext_vector_type(8)));

#define LB 4
#define LC 512
#define LT 1024
#define LF 2048
#define LH 8
#define LKC 64
#define NLAYER 6

static __device__ __forceinline__ v16h mk16(h8 lo, h8 hi) {
    return __builtin_shufflevector(lo, hi, 0, 1, 2, 3, 4, 5, 6, 7, 8, 9, 10, 11, 12, 13, 14, 15);
}
static __device__ __forceinline__ v8f wmma_f16(v16h a, v16h b, v8f c) {
    return __builtin_amdgcn_wmma_f32_16x16x32_f16(false, a, false, b, (short)0, c, false, false);
}

// ---------------- elementwise helpers ----------------

__global__ void k_cvt(const float* __restrict__ s, _Float16* __restrict__ d, int n) {
    int i = blockIdx.x * blockDim.x + threadIdx.x;
    if (i < n) d[i] = (_Float16)s[i];
}

// src: (M, K, 3) fp32  ->  dst: (3, M, K) fp16 (tap-major so GEMM A loads are contiguous)
__global__ void k_cvt_tap(const float* __restrict__ s, _Float16* __restrict__ d, int M, int K) {
    int i = blockIdx.x * blockDim.x + threadIdx.x;
    if (i >= M * K * 3) return;
    int j = i % 3;
    int k = (i / 3) % K;
    int m = i / (3 * K);
    d[((size_t)j * M + m) * K + k] = (_Float16)s[i];
}

__global__ void k_init(const float* __restrict__ x, const float* __restrict__ mask,
                       float* __restrict__ xcur, _Float16* __restrict__ xh) {
    int i = blockIdx.x * blockDim.x + threadIdx.x;
    if (i >= LB * LC * LT) return;
    int t = i % LT;
    int b = i / (LC * LT);
    float m = mask[b * LT + t];
    float v = x[i];
    xcur[i] = v;
    xh[i] = (_Float16)(v * m);
}

__global__ void k_final(const float* __restrict__ xcur, const float* __restrict__ mask,
                        float* __restrict__ out) {
    int i = blockIdx.x * blockDim.x + threadIdx.x;
    if (i >= LB * LC * LT) return;
    int t = i % LT;
    int b = i / (LC * LT);
    out[i] = xcur[i] * mask[b * LT + t];
}

// ---------------- generic WMMA GEMM / conv1d ----------------
// Out[b][m][n] = sum_{tap,k} A[tap][m][k] * X[b][k][n + tap - ntaps/2]   (+bias, relu, mask)
// A: (ntaps, M, K) fp16 row-major. X: (B, K, N) fp16.
// Block = 4 waves; block tile 128(M) x 64(N); wave tile 32(M) x 64(N).
// LDS holds a transposed (64 + 2*pad)-column x 32-k tile, staged ONCE per k-step and
// shared by all taps (B-fragment reads shift by `tap` rows). Each B fragment feeds 2 WMMAs.
__global__ void __launch_bounds__(128)
k_gemm(const _Float16* __restrict__ A, const _Float16* __restrict__ X,
       const float* __restrict__ bias, float* __restrict__ outF,
       _Float16* __restrict__ outH, const float* __restrict__ mask,
       int M, int N, int Kd, int ntaps, int relu) {
    __shared__ __align__(128) _Float16 sXT[66 * 32];  // [n_local][k_local] transposed tile
    int tid = threadIdx.x;
    int lane = tid & 31;
    int wave = tid >> 5;
    int b = blockIdx.z;
    int m0 = blockIdx.x * 128 + wave * 32;
    int n0 = blockIdx.y * 64;
    const _Float16* Xb = X + (size_t)b * Kd * N;
    int nl = lane & 15;
    int kb = (lane < 16) ? 0 : 8;
    int ldr = tid >> 2;           // staging row (k_local 0..31)
    int cseg = (tid & 3) * 16;    // staging col segment
    int pad = ntaps >> 1;         // 0 (1x1) or 1 (conv3)

    v8f acc[2][4] = {};

    for (int k0 = 0; k0 < Kd; k0 += 32) {
        __syncthreads();
        {   // stage interior 64 columns (always aligned, always in range)
            const _Float16* src = Xb + (size_t)(k0 + ldr) * N + n0 + cseg;
            h8 lo = *(const h8*)(src);
            h8 hi = *(const h8*)(src + 8);
#pragma unroll
            for (int i = 0; i < 8; ++i) {
                sXT[(pad + cseg + i) * 32 + ldr] = lo[i];
                sXT[(pad + cseg + 8 + i) * 32 + ldr] = hi[i];
            }
        }
        if (pad) {  // halo columns n0-1 and n0+64 (branch-free clamped loads)
            if (tid < 32) {
                int col = n0 - 1;
                const _Float16* src = Xb + (size_t)(k0 + tid) * N;
                sXT[tid] = (col >= 0) ? src[col] : (_Float16)0.f;      // colLocal 0
            } else if (tid < 64) {
                int col = n0 + 64;
                int kr = tid - 32;
                const _Float16* src = Xb + (size_t)(k0 + kr) * N;
                sXT[65 * 32 + kr] = (col < N) ? src[col] : (_Float16)0.f;  // colLocal 65
            }
        }
        __syncthreads();
        for (int tap = 0; tap < ntaps; ++tap) {
            const _Float16* ap0 = A + (size_t)tap * M * Kd + (size_t)(m0 + nl) * Kd + k0 + kb;
            const _Float16* ap1 = ap0 + (size_t)16 * Kd;
            __builtin_prefetch(ap0 + 32, 0, 1);  // next k-block of A -> global_prefetch_b8
            v16h a0 = mk16(*(const h8*)ap0, *(const h8*)(ap0 + 16));
            v16h a1 = mk16(*(const h8*)ap1, *(const h8*)(ap1 + 16));
            v16h bf[4];
#pragma unroll
            for (int j = 0; j < 4; ++j) {
                const _Float16* bp = &sXT[(tap + j * 16 + nl) * 32 + kb];
                bf[j] = mk16(*(const h8*)bp, *(const h8*)(bp + 16));
            }
#pragma unroll
            for (int j = 0; j < 4; ++j) {
                acc[0][j] = wmma_f16(a0, bf[j], acc[0][j]);
                acc[1][j] = wmma_f16(a1, bf[j], acc[1][j]);
            }
        }
    }
    // epilogue: C/D layout -> lane holds col n, VGPR r holds row r (+8 for upper lanes)
#pragma unroll
    for (int half = 0; half < 2; ++half) {
#pragma unroll
        for (int j = 0; j < 4; ++j) {
            int n = n0 + j * 16 + nl;
            float mv = mask ? mask[b * N + n] : 1.f;
#pragma unroll
            for (int r = 0; r < 8; ++r) {
                int m = m0 + half * 16 + r + ((lane < 16) ? 0 : 8);
                float v = acc[half][j][r] + bias[m];
                if (relu) v = fmaxf(v, 0.f);
                v *= mv;
                size_t oi = ((size_t)b * M + m) * N + n;
                if (outF) outF[oi] = v;
                if (outH) outH[oi] = (_Float16)v;
            }
        }
    }
}

// ---------------- RoPE + layout pack ----------------
// qf/kf/vf: (B,C,T) fp32 -> qh/kh: (B,H,T,64) fp16 (rope applied), vT: (B,H,64,T) fp16
__global__ void k_rope_pack(const float* __restrict__ qf, const float* __restrict__ kf,
                            const float* __restrict__ vf, _Float16* __restrict__ qh,
                            _Float16* __restrict__ kh, _Float16* __restrict__ vT) {
    int i = blockIdx.x * blockDim.x + threadIdx.x;
    if (i >= LB * LC * LT) return;
    int t = i % LT;
    int c = (i / LT) % LC;
    int b = i / (LC * LT);
    int h = c >> 6;
    int cc = c & 63;
    float qv = qf[i], kv = kf[i];
    if (cc < 32) {
        float th = powf(10000.f, -(float)(cc & 15) * (1.f / 16.f));
        float ang = (float)t * th;
        float cs = cosf(ang), sn = sinf(ang);
        int pc = (cc < 16) ? (cc + 16) : (cc - 16);
        float sgn = (cc < 16) ? -1.f : 1.f;
        size_t pidx = ((size_t)b * LC + (h * 64 + pc)) * LT + t;
        qv = qv * cs + sgn * qf[pidx] * sn;
        kv = kv * cs + sgn * kf[pidx] * sn;
    }
    size_t oq = (((size_t)(b * LH + h) * LT) + t) * LKC + cc;
    qh[oq] = (_Float16)qv;
    kh[oq] = (_Float16)kv;
    size_t ov = (((size_t)(b * LH + h) * LKC) + cc) * LT + t;
    vT[ov] = (_Float16)vf[i];
}

// ---------------- flash attention (WMMA scores + WMMA PV, online softmax) ----------------
__global__ void __launch_bounds__(128)
k_attn(const _Float16* __restrict__ qh, const _Float16* __restrict__ kh,
       const _Float16* __restrict__ vT, const float* __restrict__ mask,
       _Float16* __restrict__ oh) {
    __shared__ __align__(128) _Float16 sP[4][16][32];
    int lane = threadIdx.x & 31;
    int wave = threadIdx.x >> 5;
    int b = blockIdx.z, h = blockIdx.y;
    int t0 = (blockIdx.x * 4 + wave) * 16;
    size_t bh = (size_t)(b * LH + h);
    const _Float16* qb = qh + bh * LT * LKC;
    const _Float16* kb = kh + bh * LT * LKC;
    const _Float16* vb = vT + bh * LKC * LT;
    int nl = lane & 15;
    int kbs = (lane < 16) ? 0 : 8;

    const _Float16* qp = qb + (size_t)(t0 + nl) * LKC + kbs;
    v16h aq0 = mk16(*(const h8*)qp, *(const h8*)(qp + 16));
    v16h aq1 = mk16(*(const h8*)(qp + 32), *(const h8*)(qp + 48));

    v8f acc[4] = {};
    float mrun[8], lrun[8];
#pragma unroll
    for (int r = 0; r < 8; ++r) { mrun[r] = -1e30f; lrun[r] = 0.f; }
    const float scale = 0.125f;  // 1/sqrt(64)

    for (int s0 = 0; s0 < LT; s0 += 32) {
        const _Float16* kp0 = kb + (size_t)(s0 + nl) * LKC + kbs;
        const _Float16* kp1 = kb + (size_t)(s0 + 16 + nl) * LKC + kbs;
        v16h bk0l = mk16(*(const h8*)kp0, *(const h8*)(kp0 + 16));
        v16h bk0h = mk16(*(const h8*)(kp0 + 32), *(const h8*)(kp0 + 48));
        v16h bk1l = mk16(*(const h8*)kp1, *(const h8*)(kp1 + 16));
        v16h bk1h = mk16(*(const h8*)(kp1 + 32), *(const h8*)(kp1 + 48));
        v8f s0t = {}, s1t = {};
        s0t = wmma_f16(aq0, bk0l, s0t);
        s0t = wmma_f16(aq1, bk0h, s0t);
        s1t = wmma_f16(aq0, bk1l, s1t);
        s1t = wmma_f16(aq1, bk1h, s1t);

        float km0 = mask[b * LT + s0 + nl];
        float km1 = mask[b * LT + s0 + 16 + nl];
        float p0[8], p1[8], mnew[8];
#pragma unroll
        for (int r = 0; r < 8; ++r) {
            p0[r] = (km0 > 0.f) ? (s0t[r] * scale) : -1e4f;
            p1[r] = (km1 > 0.f) ? (s1t[r] * scale) : -1e4f;
            float v = fmaxf(p0[r], p1[r]);
#pragma unroll
            for (int off = 1; off < 16; off <<= 1) v = fmaxf(v, __shfl_xor(v, off, 16));
            mnew[r] = v;
        }
#pragma unroll
        for (int r = 0; r < 8; ++r) {
            float mt = fmaxf(mrun[r], mnew[r]);
            float alpha = __expf(mrun[r] - mt);
            mrun[r] = mt;
            p0[r] = __expf(p0[r] - mt);
            p1[r] = __expf(p1[r] - mt);
            float rs = p0[r] + p1[r];
#pragma unroll
            for (int off = 1; off < 16; off <<= 1) rs += __shfl_xor(rs, off, 16);
            lrun[r] = lrun[r] * alpha + rs;
#pragma unroll
            for (int j = 0; j < 4; ++j) acc[j][r] *= alpha;
            int row = (lane < 16) ? r : (r + 8);
            sP[wave][row][nl] = (_Float16)p0[r];
            sP[wave][row][16 + nl] = (_Float16)p1[r];
        }
        __syncthreads();
        const _Float16* pp = &sP[wave][nl][kbs];
        v16h ap = mk16(*(const h8*)pp, *(const h8*)(pp + 16));
        v16h bv[4];
#pragma unroll
        for (int j = 0; j < 4; ++j) {
            const _Float16* vp = vb + (size_t)(j * 16 + nl) * LT + s0 + kbs;
            bv[j] = mk16(*(const h8*)vp, *(const h8*)(vp + 16));
        }
#pragma unroll
        for (int j = 0; j < 4; ++j) acc[j] = wmma_f16(ap, bv[j], acc[j]);
        __syncthreads();
    }
#pragma unroll
    for (int r = 0; r < 8; ++r) {
        float inv = (lrun[r] > 0.f) ? (1.f / lrun[r]) : 0.f;
        int t = t0 + r + ((lane < 16) ? 0 : 8);
#pragma unroll
        for (int j = 0; j < 4; ++j) {
            int cc = j * 16 + nl;
            oh[((size_t)b * LC + (h * LKC + cc)) * LT + t] = (_Float16)(acc[j][r] * inv);
        }
    }
}

// ---------------- residual + LayerNorm over channels (wave per token) ----------------
__global__ void k_ln(const float* __restrict__ resid, const float* __restrict__ proj,
                     const float* __restrict__ mask, const float* __restrict__ gamma,
                     const float* __restrict__ beta, float* __restrict__ xcur,
                     _Float16* __restrict__ xh, int maskResid) {
    int lane = threadIdx.x & 31;
    int wid = (blockIdx.x * blockDim.x + threadIdx.x) >> 5;  // token id
    int b = wid / LT, t = wid % LT;
    float mval = mask[b * LT + t];
    float rm = maskResid ? mval : 1.f;
    float v[16];
    float s = 0.f;
#pragma unroll
    for (int i = 0; i < 16; ++i) {
        int c = lane + i * 32;
        size_t idx = ((size_t)b * LC + c) * LT + t;
        float u = resid[idx] * rm + proj[idx];
        v[i] = u;
        s += u;
    }
#pragma unroll
    for (int off = 1; off < 32; off <<= 1) s += __shfl_xor(s, off, 32);
    float mean = s * (1.f / LC);
    float var = 0.f;
#pragma unroll
    for (int i = 0; i < 16; ++i) { float d = v[i] - mean; var += d * d; }
#pragma unroll
    for (int off = 1; off < 32; off <<= 1) var += __shfl_xor(var, off, 32);
    float inv = rsqrtf(var * (1.f / LC) + 1e-4f);
#pragma unroll
    for (int i = 0; i < 16; ++i) {
        int c = lane + i * 32;
        size_t idx = ((size_t)b * LC + c) * LT + t;
        float y = (v[i] - mean) * inv * gamma[c] + beta[c];
        xcur[idx] = y;
        xh[idx] = (_Float16)(y * mval);
    }
}

// ---------------- host orchestration ----------------

extern "C" void kernel_launch(void* const* d_in, const int* in_sizes, int n_in,
                              void* d_out, int out_size, void* d_ws, size_t ws_size,
                              hipStream_t stream) {
    (void)in_sizes; (void)n_in; (void)out_size; (void)ws_size;
    const float* x   = (const float*)d_in[0];
    const float* xm  = (const float*)d_in[1];
    const float* Wq  = (const float*)d_in[2];
    const float* bq  = (const float*)d_in[3];
    const float* Wk  = (const float*)d_in[4];
    const float* bk  = (const float*)d_in[5];
    const float* Wv  = (const float*)d_in[6];
    const float* bv  = (const float*)d_in[7];
    const float* Wo  = (const float*)d_in[8];
    const float* bo  = (const float*)d_in[9];
    const float* g1  = (const float*)d_in[10];
    const float* be1 = (const float*)d_in[11];
    const float* W1  = (const float*)d_in[12];
    const float* c1  = (const float*)d_in[13];
    const float* W2  = (const float*)d_in[14];
    const float* c2  = (const float*)d_in[15];
    const float* g2  = (const float*)d_in[16];
    const float* be2 = (const float*)d_in[17];

    char* p = (char*)d_ws;
    auto take = [&](size_t bytes) -> char* {
        char* r = p;
        p += (bytes + 255) & ~(size_t)255;
        return r;
    };
    const size_t NE = (size_t)LB * LC * LT;
    float*    xcur = (float*)take(NE * 4);
    _Float16* xh   = (_Float16*)take(NE * 2);
    float*    qf   = (float*)take(NE * 4);
    float*    kf   = (float*)take(NE * 4);
    float*    vf   = (float*)take(NE * 4);
    _Float16* qh2  = (_Float16*)take(NE * 2);
    _Float16* kh2  = (_Float16*)take(NE * 2);
    _Float16* vTb  = (_Float16*)take(NE * 2);
    _Float16* ohb  = (_Float16*)take(NE * 2);
    _Float16* hh   = (_Float16*)take((size_t)LB * LF * LT * 2);
    _Float16* whq  = (_Float16*)take((size_t)LC * LC * 2);
    _Float16* whk  = (_Float16*)take((size_t)LC * LC * 2);
    _Float16* whv  = (_Float16*)take((size_t)LC * LC * 2);
    _Float16* who  = (_Float16*)take((size_t)LC * LC * 2);
    _Float16* w1h  = (_Float16*)take((size_t)3 * LF * LC * 2);
    _Float16* w2h  = (_Float16*)take((size_t)3 * LC * LF * 2);

    const int NEL = LB * LC * LT;
    k_init<<<(NEL + 255) / 256, 256, 0, stream>>>(x, xm, xcur, xh);

    for (int l = 0; l < NLAYER; ++l) {
        const int cc2 = LC * LC;
        k_cvt<<<(cc2 + 255) / 256, 256, 0, stream>>>(Wq + (size_t)l * cc2, whq, cc2);
        k_cvt<<<(cc2 + 255) / 256, 256, 0, stream>>>(Wk + (size_t)l * cc2, whk, cc2);
        k_cvt<<<(cc2 + 255) / 256, 256, 0, stream>>>(Wv + (size_t)l * cc2, whv, cc2);
        k_cvt<<<(cc2 + 255) / 256, 256, 0, stream>>>(Wo + (size_t)l * cc2, who, cc2);
        const int n1 = LF * LC * 3;
        k_cvt_tap<<<(n1 + 255) / 256, 256, 0, stream>>>(W1 + (size_t)l * n1, w1h, LF, LC);
        k_cvt_tap<<<(n1 + 255) / 256, 256, 0, stream>>>(W2 + (size_t)l * n1, w2h, LC, LF);

        dim3 gq(LC / 128, LT / 64, LB);
        k_gemm<<<gq, 128, 0, stream>>>(whq, xh, bq + l * LC, qf, nullptr, nullptr, LC, LT, LC, 1, 0);
        k_gemm<<<gq, 128, 0, stream>>>(whk, xh, bk + l * LC, kf, nullptr, nullptr, LC, LT, LC, 1, 0);
        k_gemm<<<gq, 128, 0, stream>>>(whv, xh, bv + l * LC, vf, nullptr, nullptr, LC, LT, LC, 1, 0);

        k_rope_pack<<<(NEL + 255) / 256, 256, 0, stream>>>(qf, kf, vf, qh2, kh2, vTb);

        dim3 ga(LT / 64, LH, LB);
        k_attn<<<ga, 128, 0, stream>>>(qh2, kh2, vTb, xm, ohb);

        k_gemm<<<gq, 128, 0, stream>>>(who, ohb, bo + l * LC, qf, nullptr, nullptr, LC, LT, LC, 1, 0);
        k_ln<<<(LB * LT) / 4, 128, 0, stream>>>(xcur, qf, xm, g1 + l * LC, be1 + l * LC, xcur, xh, 1);

        dim3 gc1(LF / 128, LT / 64, LB);
        k_gemm<<<gc1, 128, 0, stream>>>(w1h, xh, c1 + l * LF, nullptr, hh, xm, LF, LT, LC, 3, 1);
        dim3 gc2(LC / 128, LT / 64, LB);
        k_gemm<<<gc2, 128, 0, stream>>>(w2h, hh, c2 + l * LC, qf, nullptr, xm, LC, LT, LF, 3, 0);

        k_ln<<<(LB * LT) / 4, 128, 0, stream>>>(xcur, qf, xm, g2 + l * LC, be2 + l * LC, xcur, xh, 0);
    }

    k_final<<<(NEL + 255) / 256, 256, 0, stream>>>(xcur, xm, (float*)d_out);
}
